// MoERegressor_29738353558155
// MI455X (gfx1250) — matrix operand order repaired and checked by reading, not compiled, not run on hardware
//
#include <hip/hip_runtime.h>
#include <hip/hip_bf16.h>

// ---------------- problem constants ----------------
constexpr int B_ = 16384;
constexpr int D_ = 2048;
constexpr int O_ = 1024;
constexpr int E_ = 8;

// ---------------- tiling ----------------
constexpr int TT    = 128;  // token-tile rows
constexpr int OTILE = 128;  // output-column tile
constexpr int KT    = 64;   // K chunk per LDS stage
constexpr int LDA   = 72;   // padded LDS row stride (bf16 elems)
constexpr int LDB   = 72;
constexpr int NCHUNK = D_ / KT;   // 32

// ---------------- d_out layout (elements) ----------------
constexpr size_t OFF_PRED   = 0;
constexpr size_t OFF_LOGITS = (size_t)B_ * O_;
constexpr size_t OFF_SPARSE = OFF_LOGITS + (size_t)B_ * E_;
constexpr size_t OFF_IDX    = OFF_SPARSE + (size_t)B_ * E_;
constexpr size_t OFF_VALS   = OFF_IDX + (size_t)B_ * 2;

// ---------------- ws layout (bytes) ----------------
constexpr size_t WS_XBF    = 0;                                    // B*D bf16
constexpr size_t WS_WEBT   = WS_XBF + (size_t)B_ * D_ * 2;         // E*O*D bf16 (transposed!)
constexpr size_t WS_COUNTS = WS_WEBT + (size_t)E_ * D_ * O_ * 2;   // E ints (padded)
constexpr size_t WS_TOKID  = WS_COUNTS + 256;                      // E*B ints
constexpr size_t WS_TOKW   = WS_TOKID + (size_t)E_ * B_ * 4;       // E*B floats
constexpr size_t WS_WPAIR  = WS_TOKW + (size_t)E_ * B_ * 4;        // B*2 floats

typedef __attribute__((ext_vector_type(8)))  float  v8f;
typedef __attribute__((ext_vector_type(16))) __bf16 v16bf;

union ABfrag { uint4 u[2]; v16bf v; };

// ---- CDNA5 async global->LDS copy (16B per lane), ASYNCcnt-tracked ----
// LDS addr from vdst, global addr from vaddr; the imm offset is added to both
#define ASYNC_COPY64B(ldsOff, gaddr)                                                                        \
    do {                                                                                                    \
        asm volatile("global_load_async_to_lds_b128 %0, %1, off"           :: "v"(ldsOff), "v"(gaddr) : "memory"); \
        asm volatile("global_load_async_to_lds_b128 %0, %1, off offset:16" :: "v"(ldsOff), "v"(gaddr) : "memory"); \
        asm volatile("global_load_async_to_lds_b128 %0, %1, off offset:32" :: "v"(ldsOff), "v"(gaddr) : "memory"); \
        asm volatile("global_load_async_to_lds_b128 %0, %1, off offset:48" :: "v"(ldsOff), "v"(gaddr) : "memory"); \
    } while (0)

#if __has_builtin(__builtin_amdgcn_s_wait_asynccnt)
#define WAIT_ASYNC(n) __builtin_amdgcn_s_wait_asynccnt(n)
#else
#define WAIT_ASYNC(n) asm volatile("s_wait_asynccnt %0" :: "i"(n) : "memory")
#endif

static __device__ __forceinline__ unsigned short f2bf(float f) {
    unsigned int u = __float_as_uint(f);
    unsigned int r = u + 0x7FFFu + ((u >> 16) & 1u);   // round-to-nearest-even
    if ((u & 0x7F800000u) == 0x7F800000u) r = u;       // inf / nan: truncate
    return (unsigned short)(r >> 16);
}

// ---------------- kernel: zero per-expert counters ----------------
__global__ void moe_zero_counts(int* counts) {
    if (threadIdx.x < E_) counts[threadIdx.x] = 0;
}

// ---------------- kernel: f32 -> bf16 bulk convert (x) ----------------
__global__ void moe_cvt_bf16(const float* __restrict__ src,
                             unsigned short* __restrict__ dst, int n) {
    int i = blockIdx.x * 256 + threadIdx.x;           // one float4 per thread
    if (i * 4 >= n) return;
    float4 f = reinterpret_cast<const float4*>(src)[i];
    ushort4 o;
    o.x = f2bf(f.x); o.y = f2bf(f.y); o.z = f2bf(f.z); o.w = f2bf(f.w);
    reinterpret_cast<ushort4*>(dst)[i] = o;
}

// ---------------- kernel: We [e][k][o] f32 -> WeT [e][o][k] bf16 ----------------
// 32x32 tile transpose through LDS; grid = E * (D/32) * (O/32)
__global__ __launch_bounds__(256) void moe_cvt_we_T(const float* __restrict__ We,
                                                    unsigned short* __restrict__ dstT) {
    __shared__ unsigned short t[32][33];
    int bx  = blockIdx.x;
    int e   = bx >> 11;                  // / ((D/32)*(O/32)) = /2048
    int rem = bx & 2047;
    int kt  = (rem >> 5) * 32;           // k-tile base (64 tiles)
    int ot  = (rem & 31) * 32;           // o-tile base (32 tiles)

    int tid = threadIdx.x;
    int r  = tid >> 3;                   // 0..31
    int c4 = (tid & 7) * 4;              // 0,4,...,28

    float4 f = *reinterpret_cast<const float4*>(
        We + ((size_t)e * D_ + kt + r) * O_ + ot + c4);
    t[r][c4 + 0] = f2bf(f.x);
    t[r][c4 + 1] = f2bf(f.y);
    t[r][c4 + 2] = f2bf(f.z);
    t[r][c4 + 3] = f2bf(f.w);
    __syncthreads();

    ushort4 o4;
    o4.x = t[c4 + 0][r];
    o4.y = t[c4 + 1][r];
    o4.z = t[c4 + 2][r];
    o4.w = t[c4 + 3][r];
    *reinterpret_cast<ushort4*>(
        dstT + ((size_t)e * O_ + ot + r) * D_ + kt + c4) = o4;
}

// ---------------- kernel: gating (one wave32 per token) ----------------
__global__ __launch_bounds__(256) void moe_gating(
    const float* __restrict__ x, const float* __restrict__ Wg,
    const float* __restrict__ bg, float* __restrict__ out,
    int* __restrict__ counts, int* __restrict__ tokId,
    float* __restrict__ tokW, float* __restrict__ wpair) {

    int wave = threadIdx.x >> 5;
    int lane = threadIdx.x & 31;
    int b = blockIdx.x * 8 + wave;

    const float* xr = x + (size_t)b * D_;
    float acc[E_];
#pragma unroll
    for (int e = 0; e < E_; ++e) acc[e] = 0.0f;

    for (int d = lane; d < D_; d += 32) {
        float xv = xr[d];
        const float4* w4 = reinterpret_cast<const float4*>(Wg + (size_t)d * E_);
        float4 a = w4[0], c = w4[1];
        acc[0] += xv * a.x; acc[1] += xv * a.y; acc[2] += xv * a.z; acc[3] += xv * a.w;
        acc[4] += xv * c.x; acc[5] += xv * c.y; acc[6] += xv * c.z; acc[7] += xv * c.w;
    }
#pragma unroll
    for (int e = 0; e < E_; ++e)
#pragma unroll
        for (int off = 16; off > 0; off >>= 1)
            acc[e] += __shfl_xor(acc[e], off, 32);

    if (lane == 0) {
        float lg[E_], p[E_];
        float m = -1e30f;
#pragma unroll
        for (int e = 0; e < E_; ++e) { lg[e] = acc[e] + bg[e]; m = fmaxf(m, lg[e]); }
        float s = 0.0f;
#pragma unroll
        for (int e = 0; e < E_; ++e) { p[e] = expf(lg[e] - m); s += p[e]; }
        float inv = 1.0f / s;
#pragma unroll
        for (int e = 0; e < E_; ++e) p[e] *= inv;

        // top-2, ties -> lowest index (jax.lax.top_k semantics)
        int i0 = 0;
#pragma unroll
        for (int e = 1; e < E_; ++e) if (p[e] > p[i0]) i0 = e;
        int i1 = -1;
#pragma unroll
        for (int e = 0; e < E_; ++e)
            if (e != i0 && (i1 < 0 || p[e] > p[i1])) i1 = e;

        float tv0 = p[i0], tv1 = p[i1];
        float den = tv0 + tv1 + 1e-8f;
        float w0 = tv0 / den, w1 = tv1 / den;

        float* lgo = out + OFF_LOGITS + (size_t)b * E_;
        float* spo = out + OFF_SPARSE + (size_t)b * E_;
#pragma unroll
        for (int e = 0; e < E_; ++e) {
            lgo[e] = lg[e];
            spo[e] = (e == i0) ? w0 : ((e == i1) ? w1 : 0.0f);
        }
        int*   io = reinterpret_cast<int*>(out + OFF_IDX) + b * 2;
        float* vo = out + OFF_VALS + (size_t)b * 2;
        io[0] = i0; io[1] = i1;
        vo[0] = tv0; vo[1] = tv1;
        wpair[b * 2 + 0] = w0;
        wpair[b * 2 + 1] = w1;

        int p0 = atomicAdd(&counts[i0], 1);
        tokId[i0 * B_ + p0] = b;  tokW[i0 * B_ + p0] = w0;
        int p1 = atomicAdd(&counts[i1], 1);
        tokId[i1 * B_ + p1] = b;  tokW[i1 * B_ + p1] = w1;
    }
}

// ---------------- kernel: pred = w0*be[e0] + w1*be[e1] ----------------
__global__ void moe_pred_init(const float* __restrict__ be,
                              const int* __restrict__ topIdx,
                              const float* __restrict__ wpair,
                              float* __restrict__ pred) {
    int idx = blockIdx.x * 256 + threadIdx.x;
    int b = idx >> 10;           // / O_
    int o = idx & (O_ - 1);
    int i0 = topIdx[b * 2 + 0], i1 = topIdx[b * 2 + 1];
    float w0 = wpair[b * 2 + 0], w1 = wpair[b * 2 + 1];
    pred[(size_t)idx] = w0 * be[(size_t)i0 * O_ + o] + w1 * be[(size_t)i1 * O_ + o];
}

// ---------------- kernel: grouped expert GEMM (bf16 WMMA, async double-buffer) ----------------
__global__ __launch_bounds__(256, 1) void moe_expert_gemm(
    const unsigned short* __restrict__ xbf,
    const unsigned short* __restrict__ webT,   // [e][o][k], k contiguous
    const int*   __restrict__ counts,
    const int*   __restrict__ tokId,
    const float* __restrict__ tokW,
    float* __restrict__ pred) {

    __shared__ unsigned short As[2][TT * LDA];      // [row][k]
    __shared__ unsigned short Bs[2][OTILE * LDB];   // [n][k]
    __shared__ int   tokLds[TT];
    __shared__ float wLds[TT];

    int bx = blockIdx.x;
    int e     = bx >> 10;            // / ((B/TT)*(O/OTILE)) = /1024
    int rem   = bx & 1023;
    int tTile = rem >> 3;
    int oTile = rem & 7;

    int cnt    = counts[e];
    int tStart = tTile * TT;
    if (tStart >= cnt) return;
    int oBase = oTile * OTILE;

    int tid = threadIdx.x;
    if (tid < TT) {
        int g  = tStart + tid;
        bool v = (g < cnt);
        tokLds[tid] = v ? tokId[e * B_ + g] : -1;
        wLds[tid]   = v ? tokW[e * B_ + g] : 0.0f;
    }
    __syncthreads();

    int wave = tid >> 5, lane = tid & 31;
    int mBase = (wave >> 1) * 32;     // 4 waves down
    int nBase = (wave & 1) * 64;      // 2 waves across
    int lx = lane & 15, half = lane >> 4;

    v8f acc[2][4];
#pragma unroll
    for (int mi = 0; mi < 2; ++mi)
#pragma unroll
        for (int ni = 0; ni < 4; ++ni)
            acc[mi][ni] = (v8f)(0.0f);

    // loader mapping (identical for A and B tiles): row 0..127, 32-elem segment
    int la_r   = tid >> 1;
    int la_seg = (tid & 1) * 32;
    int myTok  = tokLds[la_r];
    if (myTok < 0) myTok = 0;   // garbage rows are never written out (weight row skipped)

    const unsigned short* gArow = xbf  + (size_t)myTok * D_ + la_seg;
    const unsigned short* gBrow = webT + ((size_t)e * O_ + oBase + la_r) * D_ + la_seg;

    auto issueStage = [&](int buf, int k0) {
        unsigned int ldsA = (unsigned int)(size_t)&As[buf][la_r * LDA + la_seg];
        unsigned long long gA = (unsigned long long)(size_t)(gArow + k0);
        ASYNC_COPY64B(ldsA, gA);
        unsigned int ldsB = (unsigned int)(size_t)&Bs[buf][la_r * LDB + la_seg];
        unsigned long long gB = (unsigned long long)(size_t)(gBrow + k0);
        ASYNC_COPY64B(ldsB, gB);
    };

    // software pipeline: prefetch two chunks ahead through ASYNCcnt
    issueStage(0, 0);
    issueStage(1, KT);

    for (int c = 0; c < NCHUNK; ++c) {
        int cur = c & 1;
        if (c + 1 < NCHUNK) { WAIT_ASYNC(8); } else { WAIT_ASYNC(0); }  // chunk c landed (in-order)
        __syncthreads();

#pragma unroll
        for (int kk = 0; kk < KT; kk += 32) {
            ABfrag a[2], bf[4];
#pragma unroll
            for (int mi = 0; mi < 2; ++mi) {
                int m = mBase + mi * 16 + lx;
                // 16-bit A layout: VGPR0-3 = K[half*8 .. +7], VGPR4-7 = K[16+half*8 .. +7]
                a[mi].u[0] = *reinterpret_cast<const uint4*>(&As[cur][m * LDA + kk + half * 8]);
                a[mi].u[1] = *reinterpret_cast<const uint4*>(&As[cur][m * LDA + kk + 16 + half * 8]);
            }
#pragma unroll
            for (int ni = 0; ni < 4; ++ni) {
                int n = nBase + ni * 16 + lx;
                // 16-bit B layout: VGPR0-3 = K[half*8 .. +7], VGPR4-7 = K[16+half*8 .. +7]
                const uint4* pb = reinterpret_cast<const uint4*>(&Bs[cur][n * LDB + kk + half * 8]);
                bf[ni].u[0] = pb[0];
                bf[ni].u[1] = pb[2];
            }
#pragma unroll
            for (int mi = 0; mi < 2; ++mi)
#pragma unroll
                for (int ni = 0; ni < 4; ++ni)
                    acc[mi][ni] = __builtin_amdgcn_wmma_f32_16x16x32_bf16(
                        false, a[mi].v, false, bf[ni].v,
                        (short)0, acc[mi][ni], false, false);
        }

        __syncthreads();                      // everyone done reading buf[cur]
        int nk = (c + 2) * KT;
        if (nk < D_) issueStage(cur, nk);     // refill it two chunks ahead
    }

    // ---- epilogue: scale by routing weight, atomic-add into pred ----
    // C/D layout: lane(0-15): N=lane, VGPR v -> M=v ; lane(16-31): M=8+v, N=lane-16
#pragma unroll
    for (int mi = 0; mi < 2; ++mi) {
#pragma unroll
        for (int v = 0; v < 8; ++v) {
            int row = mBase + mi * 16 + v + 8 * half;
            int tk = tokLds[row];
            if (tk < 0) continue;
            float w = wLds[row];
            float* prow = pred + (size_t)tk * O_ + oBase + nBase + lx;
#pragma unroll
            for (int ni = 0; ni < 4; ++ni)
                unsafeAtomicAdd(prow + ni * 16, w * acc[mi][ni][v]);
        }
    }
}

// ---------------- launch ----------------
extern "C" void kernel_launch(void* const* d_in, const int* in_sizes, int n_in,
                              void* d_out, int out_size, void* d_ws, size_t ws_size,
                              hipStream_t stream) {
    const float* x  = (const float*)d_in[0];
    const float* Wg = (const float*)d_in[1];
    const float* bg = (const float*)d_in[2];
    const float* We = (const float*)d_in[3];
    const float* be = (const float*)d_in[4];
    float* out = (float*)d_out;

    char* ws = (char*)d_ws;
    unsigned short* xbf  = (unsigned short*)(ws + WS_XBF);
    unsigned short* webT = (unsigned short*)(ws + WS_WEBT);
    int*   counts = (int*)(ws + WS_COUNTS);
    int*   tokId  = (int*)(ws + WS_TOKID);
    float* tokW   = (float*)(ws + WS_TOKW);
    float* wpair  = (float*)(ws + WS_WPAIR);

    moe_zero_counts<<<1, 32, 0, stream>>>(counts);

    {
        int n = B_ * D_;
        moe_cvt_bf16<<<(n / 4 + 255) / 256, 256, 0, stream>>>(x, xbf, n);
    }
    moe_cvt_we_T<<<E_ * (D_ / 32) * (O_ / 32), 256, 0, stream>>>(We, webT);

    moe_gating<<<B_ / 8, 256, 0, stream>>>(x, Wg, bg, out, counts, tokId, tokW, wpair);

    const int* topIdx = (const int*)(out + OFF_IDX);
    moe_pred_init<<<(B_ * O_) / 256, 256, 0, stream>>>(be, topIdx, wpair, out + OFF_PRED);

    int gemmBlocks = E_ * (B_ / TT) * (O_ / OTILE);   // 8192
    moe_expert_gemm<<<gemmBlocks, 256, 0, stream>>>(xbf, webT, counts, tokId, tokW,
                                                    out + OFF_PRED);
}
